// AttentionAGaLiTeLayer_61349312856289
// MI455X (gfx1250) — matrix operand order repaired and checked by reading, not compiled, not run on hardware
//
#include <hip/hip_runtime.h>
#include <hip/hip_bf16.h>
#include <math.h>

// ---------------- problem constants ----------------
#define TSEQ 4096
#define DMODEL 1024
#define NH 8
#define HD 64
#define ETA 4
#define RR 7
#define EHD 256            // ETA*HD
#define CL 256             // chunk length for the parallel scan
#define NC (TSEQ / CL)     // 16 chunks

// output tuple offsets (floats)
#define OUT_K    (TSEQ * DMODEL)                 // 4194304
#define OUT_V    (OUT_K + RR * NH * EHD)         // +14336
#define OUT_S    (OUT_V + RR * NH * HD)          // +3584
#define OUT_TICK (OUT_S + NH * EHD)              // +2048

// ---------------- WMMA / TDM types ----------------
typedef __attribute__((ext_vector_type(16))) __bf16 v16bf;
typedef __attribute__((ext_vector_type(8)))  float  v8f;
typedef unsigned int v4u __attribute__((ext_vector_type(4)));
typedef int          v8i __attribute__((ext_vector_type(8)));
typedef int          v4i __attribute__((ext_vector_type(4)));

union Frag16 { v16bf v; unsigned int u[8]; };

__device__ __forceinline__ unsigned short f2bf(float f) {
  unsigned int u = __float_as_uint(f);
  u += 0x7FFFu + ((u >> 16) & 1u);   // round to nearest even
  return (unsigned short)(u >> 16);
}

#if defined(__HIP_DEVICE_COMPILE__) && __has_builtin(__builtin_amdgcn_tensor_load_to_lds)
#define HAS_TDM 1
#else
#define HAS_TDM 0
#endif

#if HAS_TDM
// Issue a 2-D TDM load: tile_d1 rows x tile_d0 bf16 elements (contiguous),
// row stride = stride0 elements, into contiguous LDS at lds_byte_addr.
// D# bitfields per CDNA5 ISA ch.8.
__device__ __forceinline__ void tdm_load_2d_bf16(unsigned int lds_byte_addr,
                                                 const unsigned short* gptr,
                                                 unsigned int tile_d0,
                                                 unsigned int tile_d1,
                                                 unsigned long long stride0) {
  unsigned long long ga = (unsigned long long)gptr;
  v4u g0;
  g0[0] = 1u;                                              // count=1 (valid user D#)
  g0[1] = lds_byte_addr;                                   // lds_addr
  g0[2] = (unsigned int)(ga & 0xFFFFFFFFu);                // global_addr[31:0]
  g0[3] = (unsigned int)((ga >> 32) & 0x1FFFFFFu) | (2u << 30);  // addr[56:32] | type=2
  v8i g1;
  g1[0] = (int)(1u << 16);                                 // data_size=1 -> 2 bytes
  g1[1] = (int)((tile_d0 & 0xFFFFu) << 16);                // tensor_dim0[15:0]
  g1[2] = (int)(((tile_d0 >> 16) & 0xFFFFu) | ((tile_d1 & 0xFFFFu) << 16)); // td0 hi | td1 lo
  g1[3] = (int)(((tile_d1 >> 16) & 0xFFFFu) | (tile_d0 << 16));             // td1 hi | tile_dim0
  g1[4] = (int)(tile_d1 & 0xFFFFu);                        // tile_dim1 (tile_dim2=0)
  g1[5] = (int)(stride0 & 0xFFFFFFFFu);                    // tensor_dim0_stride[31:0]
  g1[6] = (int)((stride0 >> 32) & 0xFFFFu);                // stride[47:32] (stride1=0)
  g1[7] = 0;
  v4i z4 = {0, 0, 0, 0};
#if __clang_major__ >= 23
  v8i z8 = {0, 0, 0, 0, 0, 0, 0, 0};
  __builtin_amdgcn_tensor_load_to_lds(g0, g1, z4, z4, z8, 0);
#else
  __builtin_amdgcn_tensor_load_to_lds(g0, g1, z4, z4, 0);
#endif
}
#endif

// =====================================================================
// f32 -> bf16 conversion (vectorized, 8 elems/thread); n % 8 == 0
// =====================================================================
__global__ void __launch_bounds__(256)
conv_bf16(const float* __restrict__ src, unsigned short* __restrict__ dst, int n) {
  int i = (blockIdx.x * 256 + threadIdx.x) * 8;
  if (i >= n) return;
  float4 a = *(const float4*)(src + i);
  float4 b = *(const float4*)(src + i + 4);
  union { unsigned short us[8]; uint4 u; } o;
  o.us[0] = f2bf(a.x); o.us[1] = f2bf(a.y); o.us[2] = f2bf(a.z); o.us[3] = f2bf(a.w);
  o.us[4] = f2bf(b.x); o.us[5] = f2bf(b.y); o.us[6] = f2bf(b.z); o.us[7] = f2bf(b.w);
  *(uint4*)(dst + i) = o.u;
}

// =====================================================================
// f32 (K x N) -> bf16 transposed (N x K), LDS 32x32 tile transpose
// =====================================================================
__global__ void __launch_bounds__(256)
conv_bf16_t(const float* __restrict__ src, unsigned short* __restrict__ dst,
            int K, int N) {
  __shared__ unsigned short tile[32][33];
  int k0 = blockIdx.y * 32, n0 = blockIdx.x * 32;
  int tx = threadIdx.x & 31, ty = threadIdx.x >> 5;   // 32 x 8
#pragma unroll
  for (int yy = 0; yy < 32; yy += 8) {
    int k = k0 + ty + yy, n = n0 + tx;
    float v = (k < K && n < N) ? src[(size_t)k * N + n] : 0.0f;
    tile[ty + yy][tx] = f2bf(v);
  }
  __syncthreads();
#pragma unroll
  for (int yy = 0; yy < 32; yy += 8) {
    int n = n0 + ty + yy, k = k0 + tx;
    if (n < N && k < K) dst[(size_t)n * K + k] = tile[tx][ty + yy];
  }
}

// =====================================================================
// bf16 WMMA GEMM:  C(MxN) = A(MxK,bf16,row) * Bt(NxK,bf16,row)^T + bias(N)
// block tile 128x128, K-step 32, 8 waves in 4(M)x2(N), each wave a 32x64
// region = 2x4 sixteen-tiles -> 8 v_wmma per K-step.
// Tiles staged via TDM (double-buffered, wave0 drives, s_wait_tensorcnt),
// last iteration peeled so the steady-state body is branch-free.
// Requires M%128==0, K%32==0; Bt must have >= ceil(N/128)*128 rows allocated.
// =====================================================================
__global__ void __launch_bounds__(256)
wmma_gemm_bf16(const unsigned short* __restrict__ A,
               const unsigned short* __restrict__ Bt,
               const float* __restrict__ bias, float* __restrict__ C,
               int M, int N, int K) {
  __shared__ unsigned short As[2][128 * 32];   // [m][k]
  __shared__ unsigned short Bs[2][128 * 32];   // [n][k]

  const int tid  = threadIdx.x;
  const int lane = tid & 31;
  const int w    = tid >> 5;
  const int wm   = w >> 1;            // 0..3  (M direction, 32 rows each)
  const int wn   = w & 1;             // 0..1  (N direction, 64 cols each)
  const int m0   = blockIdx.y * 128;
  const int n0   = blockIdx.x * 128;
  const int half = lane >> 4;
  const int lm   = lane & 15;
  const int nk   = K / 32;

  v8f acc[2][4];
#pragma unroll
  for (int i = 0; i < 2; ++i)
#pragma unroll
    for (int j = 0; j < 4; ++j)
#pragma unroll
      for (int e = 0; e < 8; ++e) acc[i][j][e] = 0.0f;

#if HAS_TDM
  unsigned int asAddr[2], bsAddr[2];
  asAddr[0] = (unsigned int)(unsigned long long)(void*)&As[0][0];
  asAddr[1] = (unsigned int)(unsigned long long)(void*)&As[1][0];
  bsAddr[0] = (unsigned int)(unsigned long long)(void*)&Bs[0][0];
  bsAddr[1] = (unsigned int)(unsigned long long)(void*)&Bs[1][0];
  if (tid < 32) {   // wave 0 drives the DMA
    tdm_load_2d_bf16(asAddr[0], A + (size_t)m0 * K, 32, 128, (unsigned long long)K);
    tdm_load_2d_bf16(bsAddr[0], Bt + (size_t)n0 * K, 32, 128, (unsigned long long)K);
  }
#endif

  // fragment compute helper expressed inline per buffer
#define GEMM_COMPUTE(buf)                                                          \
  {                                                                                \
    Frag16 a[2], b[4];                                                             \
    _Pragma("unroll")                                                              \
    for (int ii = 0; ii < 2; ++ii) {                                               \
      const unsigned short* ap = &As[buf][(wm * 32 + ii * 16 + lm) * 32];          \
      _Pragma("unroll")                                                            \
      for (int v = 0; v < 8; ++v) {                                                \
        int k = (v < 4) ? (half * 8 + 2 * v) : (16 + half * 8 + 2 * (v - 4));      \
        a[ii].u[v] = *(const unsigned int*)(ap + k);                               \
      }                                                                            \
    }                                                                              \
    _Pragma("unroll")                                                              \
    for (int j = 0; j < 4; ++j) {                                                  \
      const unsigned short* bp = &Bs[buf][(wn * 64 + j * 16 + lm) * 32];           \
      _Pragma("unroll")                                                            \
      for (int v = 0; v < 8; ++v) {                                                \
        int k = (v < 4) ? (half * 8 + 2 * v) : (16 + half * 8 + 2 * (v - 4));      \
        b[j].u[v] = *(const unsigned int*)(bp + k);                                \
      }                                                                            \
    }                                                                              \
    _Pragma("unroll")                                                              \
    for (int ii = 0; ii < 2; ++ii)                                                 \
      _Pragma("unroll")                                                            \
      for (int j = 0; j < 4; ++j)                                                  \
        acc[ii][j] = __builtin_amdgcn_wmma_f32_16x16x32_bf16(                      \
            false, a[ii].v, false, b[j].v, (short)0, acc[ii][j], false, false);    \
  }

  // steady state: prefetch next buffer, wait for current, compute
  for (int i = 0; i < nk - 1; ++i) {
    const int cur = i & 1;
    __syncthreads();   // everyone done reading the buffer we are about to refill
#if HAS_TDM
    if (tid < 32) {
      int kk = (i + 1) * 32;
      tdm_load_2d_bf16(asAddr[cur ^ 1], A + (size_t)m0 * K + kk, 32, 128,
                       (unsigned long long)K);
      tdm_load_2d_bf16(bsAddr[cur ^ 1], Bt + (size_t)n0 * K + kk, 32, 128,
                       (unsigned long long)K);
      __builtin_amdgcn_s_wait_tensorcnt(2);   // current buffer's 2 loads complete
    }
#else
    {
      const int kk = i * 32;
#pragma unroll 4
      for (int idx = tid; idx < 128 * 16; idx += 256) {     // dword copies
        int m = idx >> 4, kp = (idx & 15) * 2;
        *(unsigned int*)&As[cur][m * 32 + kp] =
            *(const unsigned int*)(A + (size_t)(m0 + m) * K + kk + kp);
      }
#pragma unroll 4
      for (int idx = tid; idx < 128 * 16; idx += 256) {
        int n = idx >> 4, kp = (idx & 15) * 2;
        *(unsigned int*)&Bs[cur][n * 32 + kp] =
            *(const unsigned int*)(Bt + (size_t)(n0 + n) * K + kk + kp);
      }
    }
#endif
    __syncthreads();
    GEMM_COMPUTE(cur);
  }

  // peeled last iteration
  {
    const int cur = (nk - 1) & 1;
    __syncthreads();
#if HAS_TDM
    if (tid < 32) __builtin_amdgcn_s_wait_tensorcnt(0);
#else
    {
      const int kk = (nk - 1) * 32;
#pragma unroll 4
      for (int idx = tid; idx < 128 * 16; idx += 256) {
        int m = idx >> 4, kp = (idx & 15) * 2;
        *(unsigned int*)&As[cur][m * 32 + kp] =
            *(const unsigned int*)(A + (size_t)(m0 + m) * K + kk + kp);
      }
#pragma unroll 4
      for (int idx = tid; idx < 128 * 16; idx += 256) {
        int n = idx >> 4, kp = (idx & 15) * 2;
        *(unsigned int*)&Bs[cur][n * 32 + kp] =
            *(const unsigned int*)(Bt + (size_t)(n0 + n) * K + kk + kp);
      }
    }
#endif
    __syncthreads();
    GEMM_COMPUTE(cur);
  }
#undef GEMM_COMPUTE

  // store C (+bias).  C layout: VGPR r -> M = base + r + 8*(lane>=16), N = lane&15
#pragma unroll
  for (int i = 0; i < 2; ++i) {
    int mbase = m0 + wm * 32 + i * 16 + half * 8;
#pragma unroll
    for (int j = 0; j < 4; ++j) {
      int n = n0 + wn * 64 + j * 16 + lm;
      if (n < N) {
        float bv = bias[n];
#pragma unroll
        for (int r = 0; r < 8; ++r)
          C[(size_t)(mbase + r) * N + n] = acc[i][j][r] + bv;
      }
    }
  }
}

// =====================================================================
// Fused activation / feature map.  One thread per (t,h,eta,hd) element.
// =====================================================================
__global__ void __launch_bounds__(256)
activation_kernel(const float* __restrict__ kqv, const float* __restrict__ p,
                  const int* __restrict__ term, const float* __restrict__ tick,
                  float* __restrict__ kg, float* __restrict__ ak,
                  float* __restrict__ qf, float* __restrict__ vb,
                  float* __restrict__ av, float* __restrict__ occ) {
  size_t id = (size_t)blockIdx.x * 256 + threadIdx.x;  // < T*H*256
  int d2 = (int)(id & 255);
  int h  = (int)((id >> 8) & 7);
  int t  = (int)(id >> 11);
  int n  = d2 >> 6, d = d2 & 63;

  const float* krow = kqv + (size_t)t * 2560 + h * 320;
  const float* prow = p   + (size_t)t * 96   + h * 12;
  float nt = 1.0f - (float)term[t];

  float key = fmaxf(krow[d], 0.0f);
  float qu  = fmaxf(krow[64 + d], 0.0f);
  float p1  = fmaxf(prow[n], 0.0f);
  float p2  = fmaxf(prow[4 + n], 0.0f);
  float sg  = 1.0f / (1.0f + expf(-krow[256 + d]));
  float sp3 = 1.0f / (1.0f + expf(-prow[8 + n]));
  float gam = sg * sp3;

  size_t o = ((size_t)t * NH + h) * EHD + d2;
  kg[o] = key * p1 * gam;
  ak[o] = (1.0f - gam) * nt;
  qf[o] = qu * p2;

  if (n == 0) {
    float val = krow[128 + d];
    float sb  = 1.0f / (1.0f + expf(-krow[192 + d]));
    size_t o2 = ((size_t)t * NH + h) * HD + d;
    vb[o2] = val * sb;
    av[o2] = (1.0f - sb) * nt;
  }
  if (h == 0 && d2 < RR) {
    const float PI = 3.14159265358979323846f;
    float omega = -PI + (2.0f * PI / 6.0f) * (float)d2;
    occ[t * RR + d2] = cosf((tick[0] + (float)(t + 1)) * omega);
  }
}

// =====================================================================
// Scan pass 1: per-chunk local scan -> chunk aggregates (prod(a), end state)
// =====================================================================
__global__ void __launch_bounds__(256)
scan_pass1(const float* __restrict__ kg, const float* __restrict__ ak,
           const float* __restrict__ vb, const float* __restrict__ av,
           const float* __restrict__ occ,
           float* __restrict__ kA, float* __restrict__ kEnd,
           float* __restrict__ vA, float* __restrict__ vEnd,
           float* __restrict__ sEnd) {
  const int h = blockIdx.x & 7;
  const int c = blockIdx.x >> 3;
  const int d = threadIdx.x;
  const bool dov = d < HD;

  float kSt[RR], vSt[RR];
  float kAcc = 1.0f, vAcc = 1.0f, sSt = 0.0f;
#pragma unroll
  for (int r = 0; r < RR; ++r) { kSt[r] = 0.0f; vSt[r] = 0.0f; }

  for (int tl = 0; tl < CL; ++tl) {
    int t = c * CL + tl;
    size_t base = ((size_t)t * NH + h) * EHD + d;
    float a = ak[base], b = kg[base];
    float oc[RR];
#pragma unroll
    for (int r = 0; r < RR; ++r) oc[r] = occ[t * RR + r];
    kAcc *= a;
    sSt = sSt * a + b;
#pragma unroll
    for (int r = 0; r < RR; ++r) kSt[r] = kSt[r] * a + b * oc[r];
    if (dov) {
      size_t vi = ((size_t)t * NH + h) * HD + d;
      float a2 = av[vi], b2 = vb[vi];
      vAcc *= a2;
#pragma unroll
      for (int r = 0; r < RR; ++r) vSt[r] = vSt[r] * a2 + b2 * oc[r];
    }
  }
  size_t ki = ((size_t)c * NH + h) * EHD + d;
  kA[ki] = kAcc;
  sEnd[ki] = sSt;
#pragma unroll
  for (int r = 0; r < RR; ++r)
    kEnd[(((size_t)c * RR + r) * NH + h) * EHD + d] = kSt[r];
  if (dov) {
    size_t vi = ((size_t)c * NH + h) * HD + d;
    vA[vi] = vAcc;
#pragma unroll
    for (int r = 0; r < RR; ++r)
      vEnd[(((size_t)c * RR + r) * NH + h) * HD + d] = vSt[r];
  }
}

// =====================================================================
// Pass 2: serial scan over NC chunk aggregates -> per-chunk prefixes,
// also emits final_* outputs + new_tick.
// =====================================================================
__global__ void __launch_bounds__(256)
chunk_prefix(const float* __restrict__ kA, const float* __restrict__ kEnd,
             const float* __restrict__ vA, const float* __restrict__ vEnd,
             const float* __restrict__ sEnd,
             const float* __restrict__ k_prev, const float* __restrict__ v_prev,
             const float* __restrict__ s_prev, const float* __restrict__ tick,
             float* __restrict__ kPre, float* __restrict__ vPre,
             float* __restrict__ sPre, float* __restrict__ out) {
  int gid = blockIdx.x * 256 + threadIdx.x;
  if (gid == 0) out[OUT_TICK] = tick[0] + (float)TSEQ;

  if (gid < RR * NH * EHD) {
    int d = gid & 255, h = (gid >> 8) & 7, r = gid >> 11;
    float st = k_prev[gid];
    for (int c = 0; c < NC; ++c) {
      size_t pi = (((size_t)c * RR + r) * NH + h) * EHD + d;
      kPre[pi] = st;
      st = st * kA[((size_t)c * NH + h) * EHD + d] + kEnd[pi];
    }
    out[OUT_K + gid] = st;
  } else if (gid < RR * NH * EHD + RR * NH * HD) {
    int i = gid - RR * NH * EHD;
    int d = i & 63, h = (i >> 6) & 7, r = i >> 9;
    float st = v_prev[i];
    for (int c = 0; c < NC; ++c) {
      size_t pi = (((size_t)c * RR + r) * NH + h) * HD + d;
      vPre[pi] = st;
      st = st * vA[((size_t)c * NH + h) * HD + d] + vEnd[pi];
    }
    out[OUT_V + i] = st;
  } else if (gid < RR * NH * EHD + RR * NH * HD + NH * EHD) {
    int i = gid - RR * NH * EHD - RR * NH * HD;
    int d = i & 255, h = i >> 8;
    float st = s_prev[i];
    for (int c = 0; c < NC; ++c) {
      size_t pi = ((size_t)c * NH + h) * EHD + d;
      sPre[pi] = st;
      st = st * kA[pi] + sEnd[pi];
    }
    out[OUT_S + i] = st;
  }
}

// =====================================================================
// Pass 3: re-scan each chunk from its prefix, fused with kdq/norm
// reductions (wave32 shuffles + LDS) and the kv/attn normalization.
// =====================================================================
__global__ void __launch_bounds__(256)
scan_pass2(const float* __restrict__ kg, const float* __restrict__ ak,
           const float* __restrict__ vb, const float* __restrict__ av,
           const float* __restrict__ qf, const float* __restrict__ occ,
           const float* __restrict__ kPre, const float* __restrict__ vPre,
           const float* __restrict__ sPre, float* __restrict__ attn) {
  __shared__ float red[8][8];
  __shared__ float bc[8];
  const int h = blockIdx.x & 7;
  const int c = blockIdx.x >> 3;
  const int tid = threadIdx.x;
  const int lane = tid & 31;
  const int w = tid >> 5;
  const bool dov = tid < HD;

  float kSt[RR], vSt[RR], sSt;
#pragma unroll
  for (int r = 0; r < RR; ++r)
    kSt[r] = kPre[(((size_t)c * RR + r) * NH + h) * EHD + tid];
  sSt = sPre[((size_t)c * NH + h) * EHD + tid];
#pragma unroll
  for (int r = 0; r < RR; ++r)
    vSt[r] = dov ? vPre[(((size_t)c * RR + r) * NH + h) * HD + tid] : 0.0f;

  for (int tl = 0; tl < CL; ++tl) {
    int t = c * CL + tl;
    size_t base = ((size_t)t * NH + h) * EHD + tid;
    float a = ak[base], b = kg[base], qv = qf[base];
    float oc[RR];
#pragma unroll
    for (int r = 0; r < RR; ++r) oc[r] = occ[t * RR + r];

    sSt = sSt * a + b;
    float part[8];
    part[7] = sSt * qv;
#pragma unroll
    for (int r = 0; r < RR; ++r) {
      kSt[r] = kSt[r] * a + b * oc[r];
      part[r] = kSt[r] * qv;
    }
    if (dov) {
      size_t vi = ((size_t)t * NH + h) * HD + tid;
      float a2 = av[vi], b2 = vb[vi];
#pragma unroll
      for (int r = 0; r < RR; ++r) vSt[r] = vSt[r] * a2 + b2 * oc[r];
    }
#pragma unroll
    for (int i = 0; i < 8; ++i) {
      part[i] += __shfl_xor(part[i], 16, 32);
      part[i] += __shfl_xor(part[i], 8, 32);
      part[i] += __shfl_xor(part[i], 4, 32);
      part[i] += __shfl_xor(part[i], 2, 32);
      part[i] += __shfl_xor(part[i], 1, 32);
    }
    if (lane == 0) {
#pragma unroll
      for (int i = 0; i < 8; ++i) red[w][i] = part[i];
    }
    __syncthreads();
    if (tid < 8) {
      float s = 0.0f;
#pragma unroll
      for (int w2 = 0; w2 < 8; ++w2) s += red[w2][tid];
      bc[tid] = s;
    }
    __syncthreads();
    if (dov) {
      float norm = bc[7];
      float kv = 0.0f;
#pragma unroll
      for (int r = 0; r < RR; ++r) kv += vSt[r] * bc[r];
      attn[((size_t)t * NH + h) * HD + tid] = kv / (2.0f * RR * norm + 1e-5f);
    }
  }
}

// =====================================================================
extern "C" void kernel_launch(void* const* d_in, const int* in_sizes, int n_in,
                              void* d_out, int out_size, void* d_ws, size_t ws_size,
                              hipStream_t stream) {
  const float* inputs       = (const float*)d_in[0];
  const int*   terminations = (const int*)  d_in[1];
  const float* k_prev       = (const float*)d_in[2];
  const float* v_prev       = (const float*)d_in[3];
  const float* s_prev       = (const float*)d_in[4];
  const float* tick         = (const float*)d_in[5];
  const float* W_kqv        = (const float*)d_in[6];
  const float* b_kqv        = (const float*)d_in[7];
  const float* W_p          = (const float*)d_in[8];
  const float* b_p          = (const float*)d_in[9];
  const float* W_proj       = (const float*)d_in[10];
  const float* b_proj       = (const float*)d_in[11];
  float* out = (float*)d_out;
  float* ws  = (float*)d_ws;

  size_t off = 0;
  float* kqv  = ws + off; off += (size_t)TSEQ * 2560;
  float* pbuf = ws + off; off += (size_t)TSEQ * 96;
  float* occ  = ws + off; off += (size_t)TSEQ * RR;
  float* kg   = ws + off; off += (size_t)TSEQ * NH * EHD;
  float* ak   = ws + off; off += (size_t)TSEQ * NH * EHD;
  float* qf   = ws + off; off += (size_t)TSEQ * NH * EHD;
  float* vb   = ws + off; off += (size_t)TSEQ * NH * HD;
  float* av   = ws + off; off += (size_t)TSEQ * NH * HD;
  float* attn = ws + off; off += (size_t)TSEQ * NH * HD;
  float* kA   = ws + off; off += (size_t)NC * NH * EHD;
  float* kEnd = ws + off; off += (size_t)NC * RR * NH * EHD;
  float* kPre = ws + off; off += (size_t)NC * RR * NH * EHD;
  float* vA   = ws + off; off += (size_t)NC * NH * HD;
  float* vEnd = ws + off; off += (size_t)NC * RR * NH * HD;
  float* vPre = ws + off; off += (size_t)NC * RR * NH * HD;
  float* sEnd = ws + off; off += (size_t)NC * NH * EHD;
  float* sPre = ws + off; off += (size_t)NC * NH * EHD;
  // bf16 operand staging (ushort arrays carved out of the float workspace)
  unsigned short* Abf    = (unsigned short*)(ws + off); off += (size_t)TSEQ * DMODEL / 2;
  unsigned short* Wkqv_t = (unsigned short*)(ws + off); off += (size_t)2560 * DMODEL / 2;
  unsigned short* Wp_t   = (unsigned short*)(ws + off); off += (size_t)128 * DMODEL / 2;   // 96 padded to 128 rows
  unsigned short* Wpr_t  = (unsigned short*)(ws + off); off += (size_t)DMODEL * 512 / 2;
  unsigned short* Atbf   = (unsigned short*)(ws + off); off += (size_t)TSEQ * 512 / 2;

  // 0) convert operands to bf16 (A row-major, weights transposed to NxK)
  conv_bf16<<<(TSEQ * DMODEL) / (8 * 256), 256, 0, stream>>>(inputs, Abf, TSEQ * DMODEL);
  conv_bf16_t<<<dim3(2560 / 32, DMODEL / 32), 256, 0, stream>>>(W_kqv, Wkqv_t, DMODEL, 2560);
  conv_bf16_t<<<dim3(3, DMODEL / 32), 256, 0, stream>>>(W_p, Wp_t, DMODEL, 96);
  conv_bf16_t<<<dim3(DMODEL / 32, 512 / 32), 256, 0, stream>>>(W_proj, Wpr_t, 512, DMODEL);

  // 1) kqv = inputs @ W_kqv + b_kqv   (4096 x 2560, K=1024)  [WMMA + TDM]
  wmma_gemm_bf16<<<dim3(2560 / 128, TSEQ / 128), 256, 0, stream>>>(
      Abf, Wkqv_t, b_kqv, kqv, TSEQ, 2560, DMODEL);
  // 2) p = inputs @ W_p + b_p         (4096 x 96, K=1024)    [WMMA + TDM]
  wmma_gemm_bf16<<<dim3(1, TSEQ / 128), 256, 0, stream>>>(
      Abf, Wp_t, b_p, pbuf, TSEQ, 96, DMODEL);
  // 3) activations / feature maps / oscillators
  activation_kernel<<<TSEQ * NH, 256, 0, stream>>>(
      kqv, pbuf, terminations, tick, kg, ak, qf, vb, av, occ);
  // 4) chunked associative scan
  scan_pass1<<<NC * NH, 256, 0, stream>>>(kg, ak, vb, av, occ, kA, kEnd, vA, vEnd, sEnd);
  chunk_prefix<<<(RR * NH * EHD + RR * NH * HD + NH * EHD + 255) / 256, 256, 0, stream>>>(
      kA, kEnd, vA, vEnd, sEnd, k_prev, v_prev, s_prev, tick, kPre, vPre, sPre, out);
  scan_pass2<<<NC * NH, 256, 0, stream>>>(kg, ak, vb, av, qf, occ, kPre, vPre, sPre, attn);
  // 5) attn_out = attn @ W_proj + b_proj  (4096 x 1024, K=512)  [WMMA + TDM]
  conv_bf16<<<(TSEQ * 512) / (8 * 256), 256, 0, stream>>>(attn, Atbf, TSEQ * 512);
  wmma_gemm_bf16<<<dim3(DMODEL / 128, TSEQ / 128), 256, 0, stream>>>(
      Atbf, Wpr_t, b_proj, out, TSEQ, DMODEL, 512);
}
